// AttModel_syb_40235253629204
// MI455X (gfx1250) — compile-verified, matrix operand order checked by reading, not run on hardware
//
#include <hip/hip_runtime.h>

// ---------------------------------------------------------------------------
// dims
#define BB 32
#define LS 192
#define LQ 64
#define TT 256          // LS + LQ
#define HH 512
#define NHD 8
#define HD 64           // head dim
#define NB 6
#define DG 300
#define DM 2048
#define HF 2048
#define BT (BB*TT)      // 8192 rows
#define NEGV (-4294967295.0f)

typedef __attribute__((ext_vector_type(16))) __bf16 v16bf;
typedef __attribute__((ext_vector_type(8)))  float  v8f;

__device__ __forceinline__ __bf16 f2bf(float f) {
  union { float f; unsigned u; } a; a.f = f;
  unsigned u = (a.u + 0x7FFFu + ((a.u >> 16) & 1u)) >> 16;
  union { unsigned short s; __bf16 h; } b; b.s = (unsigned short)u;
  return b.h;
}
// pack two floats into one dword of bf16s (lo in [15:0], hi in [31:16])
__device__ __forceinline__ unsigned pk2(float lo, float hi) {
  union { float f; unsigned u; } a, b; a.f = lo; b.f = hi;
  unsigned ul = (a.u + 0x7FFFu + ((a.u >> 16) & 1u)) >> 16;
  unsigned uh = (b.u + 0x7FFFu + ((b.u >> 16) & 1u)) >> 16;
  return (uh << 16) | ul;
}

__device__ __forceinline__ v8f wmma_bf16(v16bf a, v16bf b, v8f c) {
  // 8 args: (neg_a, A, neg_b, B, c_mod, C, reuse_a, reuse_b)
  return __builtin_amdgcn_wmma_f32_16x16x32_bf16(false, a, false, b, (short)0, c, false, false);
}

// ---------------------------------------------------------------------------
// Generic GEMM: C[M,N] = act(A[M,K](lda) @ W[K,N] + bias), bf16 WMMA, f32 accum
// block = 256 threads (8 waves). WG tile 128x64. K arbitrary (guarded tail).
#define GBM 128
#define GBN 64
#define GBK 32

__global__ __launch_bounds__(256) void k_gemm_nn(
    const float* __restrict__ A, int lda,
    const float* __restrict__ W, const float* __restrict__ bias,
    float* __restrict__ C, int M, int N, int K, int relu_act)
{
  __shared__ __bf16 As[GBM * GBK];
  __shared__ __bf16 Ws[GBN * GBK];
  int tid  = threadIdx.x;
  int wave = tid >> 5, lane = tid & 31;
  int half = (lane >> 4) & 1, lm = lane & 15;
  int klo  = half * 8;
  int m0 = blockIdx.x * GBM;
  int n0 = blockIdx.y * GBN;

  v8f acc[4];
  for (int i = 0; i < 4; i++)
    for (int j = 0; j < 8; j++) acc[i][j] = 0.f;

  // fast-path thread mapping (branch-free vectorized staging)
  const int ar_  = tid >> 1;          // A: row 0..127
  const int ac16 = (tid & 1) * 16;    // A: col chunk 0 / 16
  const int wkp  = (tid >> 4) * 2;    // W: k pair base 0..30
  const int wn4  = (tid & 15) * 4;    // W: 4 consecutive n

  const bool mfull = (m0 + GBM <= M);

  for (int k0 = 0; k0 < K; k0 += GBK) {
    if (mfull && (k0 + GBK <= K)) {
      // ---- A tile: 16 floats/thread via float4, pack to bf16, uint4 LDS store
      const float4* ap = (const float4*)(A + (size_t)(m0 + ar_) * lda + k0 + ac16);
      float4 a0 = ap[0], a1 = ap[1], a2 = ap[2], a3 = ap[3];
      uint4* ad = reinterpret_cast<uint4*>(&As[ar_ * GBK + ac16]);
      ad[0] = make_uint4(pk2(a0.x, a0.y), pk2(a0.z, a0.w), pk2(a1.x, a1.y), pk2(a1.z, a1.w));
      ad[1] = make_uint4(pk2(a2.x, a2.y), pk2(a2.z, a2.w), pk2(a3.x, a3.y), pk2(a3.z, a3.w));
      // ---- W tile transposed: rows kp,kp+1 x 4 cols -> 4 packed dword stores
      const float* wp = W + (size_t)(k0 + wkp) * N + n0 + wn4;
      float4 w0 = *(const float4*)wp;
      float4 w1 = *(const float4*)(wp + N);
      unsigned* wd = reinterpret_cast<unsigned*>(Ws);
      int kh = wkp >> 1;
      wd[(wn4 + 0) * 16 + kh] = pk2(w0.x, w1.x);
      wd[(wn4 + 1) * 16 + kh] = pk2(w0.y, w1.y);
      wd[(wn4 + 2) * 16 + kh] = pk2(w0.z, w1.z);
      wd[(wn4 + 3) * 16 + kh] = pk2(w0.w, w1.w);
      if (k0 + 2 * GBK <= K)   // prefetch next W tile (global_prefetch_b8)
        __builtin_prefetch(W + (size_t)(k0 + GBK + wkp) * N + n0 + wn4, 0, 1);
    } else {
      // ---- guarded scalar staging (edge tiles only)
      {
        int c = tid & 31, r = tid >> 5;
        for (int rr = r; rr < GBM; rr += 8) {
          int row = m0 + rr, k = k0 + c;
          float v = (row < M && k < K) ? A[(size_t)row * lda + k] : 0.f;
          As[rr * GBK + c] = f2bf(v);
        }
      }
      {
        int n = tid & 63, cb = tid >> 6;
        for (int cc = cb; cc < GBK; cc += 4) {
          int k = k0 + cc;
          float v = (k < K) ? W[(size_t)k * N + n0 + n] : 0.f;
          Ws[n * GBK + cc] = f2bf(v);
        }
      }
    }
    __syncthreads();

    v16bf af;
    int arow = wave * 16 + lm;
    for (int j = 0; j < 8; j++) {
      af[j]     = As[arow * GBK + klo + j];
      af[8 + j] = As[arow * GBK + 16 + klo + j];
    }
    // one B fragment live at a time: ds_load_b128 pair -> v_wmma, x4
    for (int nt = 0; nt < 4; nt++) {
      v16bf bfv;
      int bn = nt * 16 + lm;
      for (int j = 0; j < 8; j++) {
        bfv[j]     = Ws[bn * GBK + klo + j];
        bfv[8 + j] = Ws[bn * GBK + 16 + klo + j];
      }
      acc[nt] = wmma_bf16(af, bfv, acc[nt]);
    }
    __syncthreads();
  }

  if (mfull) {
    for (int nt = 0; nt < 4; nt++) {
      int col = n0 + nt * 16 + lm;
      float bv = bias[col];
      for (int j = 0; j < 8; j++) {
        int row = m0 + wave * 16 + half * 8 + j;
        float v = acc[nt][j] + bv;
        if (relu_act) v = fmaxf(v, 0.f);
        C[(size_t)row * N + col] = v;
      }
    }
  } else {
    for (int nt = 0; nt < 4; nt++)
      for (int j = 0; j < 8; j++) {
        int row = m0 + wave * 16 + half * 8 + j;
        int col = n0 + nt * 16 + lm;
        if (row < M) {
          float v = acc[nt][j] + bias[col];
          if (relu_act) v = fmaxf(v, 0.f);
          C[(size_t)row * N + col] = v;
        }
      }
  }
}

// ---------------------------------------------------------------------------
// Encoder self-attention per (b, head, 32-q-row tile). 64 threads = 2 waves,
// each wave owns 16 q rows. Scores via WMMA, softmax in wave-private LDS,
// P@V via WMMA.
__global__ __launch_bounds__(64) void k_attn_enc(
    const float* __restrict__ Q, const float* __restrict__ Kb,
    const float* __restrict__ Vb, const float* __restrict__ km,
    const float* __restrict__ graph, float* __restrict__ O)
{
  __shared__ float S[2][16][TT];
  int tid = threadIdx.x, wave = tid >> 5, lane = tid & 31;
  int half = (lane >> 4) & 1, lm = lane & 15, klo = half * 8;
  int bid = blockIdx.x;
  int qt = bid & 7;            // 8 tiles of 32 q rows
  int h  = (bid >> 3) & 7;
  int b  = bid >> 6;
  int q0 = qt * 32 + wave * 16;
  int c0 = h * HD;
  const size_t rb = (size_t)b * TT;

  // Q fragments (two 32-wide K chunks over the 64 channels), float4 loads
  v16bf aq[2];
  for (int cc = 0; cc < 2; cc++) {
    const float4* qp = (const float4*)(Q + (rb + q0 + lm) * HH + c0 + cc * 32 + klo);
    float4 x0 = qp[0], x1 = qp[1], y0 = qp[4], y1 = qp[5];
    aq[cc][0] = f2bf(x0.x);  aq[cc][1] = f2bf(x0.y);  aq[cc][2] = f2bf(x0.z);  aq[cc][3] = f2bf(x0.w);
    aq[cc][4] = f2bf(x1.x);  aq[cc][5] = f2bf(x1.y);  aq[cc][6] = f2bf(x1.z);  aq[cc][7] = f2bf(x1.w);
    aq[cc][8] = f2bf(y0.x);  aq[cc][9] = f2bf(y0.y);  aq[cc][10] = f2bf(y0.z); aq[cc][11] = f2bf(y0.w);
    aq[cc][12] = f2bf(y1.x); aq[cc][13] = f2bf(y1.y); aq[cc][14] = f2bf(y1.z); aq[cc][15] = f2bf(y1.w);
  }

  // scores
  for (int kb = 0; kb < 16; kb++) {
    int n0 = kb * 16;
    v8f acc; for (int j = 0; j < 8; j++) acc[j] = 0.f;
    for (int cc = 0; cc < 2; cc++) {
      const float4* kp = (const float4*)(Kb + (rb + n0 + lm) * HH + c0 + cc * 32 + klo);
      float4 x0 = kp[0], x1 = kp[1], y0 = kp[4], y1 = kp[5];
      v16bf bk;
      bk[0] = f2bf(x0.x);  bk[1] = f2bf(x0.y);  bk[2] = f2bf(x0.z);  bk[3] = f2bf(x0.w);
      bk[4] = f2bf(x1.x);  bk[5] = f2bf(x1.y);  bk[6] = f2bf(x1.z);  bk[7] = f2bf(x1.w);
      bk[8] = f2bf(y0.x);  bk[9] = f2bf(y0.y);  bk[10] = f2bf(y0.z); bk[11] = f2bf(y0.w);
      bk[12] = f2bf(y1.x); bk[13] = f2bf(y1.y); bk[14] = f2bf(y1.z); bk[15] = f2bf(y1.w);
      acc = wmma_bf16(aq[cc], bk, acc);
    }
    int n = n0 + lm;
    float kmv = km[rb + n];
    for (int j = 0; j < 8; j++) {
      int r = half * 8 + j;
      int qa = q0 + r;
      float s = acc[j] * 0.125f;                      // 1/sqrt(64)
      if (kmv == 0.f) s = NEGV;
      if (graph[((size_t)b * TT + qa) * TT + n] == 0.f) s = NEGV;
      S[wave][r][n] = s;
    }
  }

  // softmax (per row, 32 lanes cooperate) + q-mask
  for (int r = 0; r < 16; r++) {
    int qa = q0 + r;
    float qmv = km[rb + qa];
    float m = -3.4e38f;
    for (int c = lane; c < TT; c += 32) m = fmaxf(m, S[wave][r][c]);
    for (int off = 16; off > 0; off >>= 1) m = fmaxf(m, __shfl_xor(m, off));
    float sum = 0.f;
    for (int c = lane; c < TT; c += 32) {
      float e = __expf(S[wave][r][c] - m);
      S[wave][r][c] = e; sum += e;
    }
    for (int off = 16; off > 0; off >>= 1) sum += __shfl_xor(sum, off);
    float sc = qmv / sum;
    for (int c = lane; c < TT; c += 32) S[wave][r][c] *= sc;
  }

  // P @ V
  v8f o[4];
  for (int i = 0; i < 4; i++) for (int j = 0; j < 8; j++) o[i][j] = 0.f;
  for (int kb = 0; kb < 8; kb++) {
    int k0 = kb * 32;
    const float4* pp = (const float4*)(&S[wave][lm][k0 + klo]);
    float4 x0 = pp[0], x1 = pp[1], y0 = pp[4], y1 = pp[5];
    v16bf ap;
    ap[0] = f2bf(x0.x);  ap[1] = f2bf(x0.y);  ap[2] = f2bf(x0.z);  ap[3] = f2bf(x0.w);
    ap[4] = f2bf(x1.x);  ap[5] = f2bf(x1.y);  ap[6] = f2bf(x1.z);  ap[7] = f2bf(x1.w);
    ap[8] = f2bf(y0.x);  ap[9] = f2bf(y0.y);  ap[10] = f2bf(y0.z); ap[11] = f2bf(y0.w);
    ap[12] = f2bf(y1.x); ap[13] = f2bf(y1.y); ap[14] = f2bf(y1.z); ap[15] = f2bf(y1.w);
    for (int ct = 0; ct < 4; ct++) {
      v16bf bv;
      int c = c0 + ct * 16 + lm;
      for (int j = 0; j < 8; j++) {
        bv[j]     = f2bf(Vb[(rb + k0 + klo + j) * HH + c]);
        bv[8 + j] = f2bf(Vb[(rb + k0 + 16 + klo + j) * HH + c]);
      }
      o[ct] = wmma_bf16(ap, bv, o[ct]);
    }
  }
  for (int ct = 0; ct < 4; ct++)
    for (int j = 0; j < 8; j++) {
      int qa = q0 + half * 8 + j;
      O[(rb + qa) * HH + c0 + ct * 16 + lm] = o[ct][j];
    }
}

// ---------------------------------------------------------------------------
// row mask: km[row] = sign(|sum_c X[row,c]|)   (wave per row)
__global__ __launch_bounds__(256) void k_rowmask(const float* __restrict__ X,
                                                 float* __restrict__ km, int rows) {
  int wave = threadIdx.x >> 5, lane = threadIdx.x & 31;
  int row = blockIdx.x * 8 + wave;
  if (row >= rows) return;
  float s = 0.f;
  for (int c = lane; c < HH; c += 32) s += X[(size_t)row * HH + c];
  for (int off = 16; off > 0; off >>= 1) s += __shfl_xor(s, off);
  if (lane == 0) km[row] = (fabsf(s) != 0.f) ? 1.f : 0.f;
}

// residual + layernorm: out = LN(a + res)*g + beta   (wave per 512-wide row)
__global__ __launch_bounds__(256) void k_add_ln(
    const float* __restrict__ a, const float* __restrict__ res,
    const float* __restrict__ g, const float* __restrict__ beta,
    float* __restrict__ out, int rows) {
  int wave = threadIdx.x >> 5, lane = threadIdx.x & 31;
  int row = blockIdx.x * 8 + wave;
  if (row >= rows) return;
  float v[16]; float s = 0.f;
  for (int i = 0; i < 16; i++) {
    int c = lane + i * 32;
    v[i] = a[(size_t)row * HH + c] + res[(size_t)row * HH + c];
    s += v[i];
  }
  for (int off = 16; off > 0; off >>= 1) s += __shfl_xor(s, off);
  float mean = s * (1.f / HH);
  float q = 0.f;
  for (int i = 0; i < 16; i++) { float d = v[i] - mean; q += d * d; }
  for (int off = 16; off > 0; off >>= 1) q += __shfl_xor(q, off);
  float rs = rsqrtf(q * (1.f / HH) + 1e-8f);
  for (int i = 0; i < 16; i++) {
    int c = lane + i * 32;
    out[(size_t)row * HH + c] = (v[i] - mean) * rs * g[c] + beta[c];
  }
}

// embedding gather: X0[b*T+t, c] = emb[idx, c]  (c < 300)
__global__ void k_gather(const int* __restrict__ syb, const int* __restrict__ qf,
                         const float* __restrict__ emb, float* __restrict__ X0) {
  int t = blockIdx.x * blockDim.x + threadIdx.x;
  if (t >= BT * DG) return;
  int row = t / DG, c = t - row * DG;
  int b = row >> 8, tt = row & 255;
  int id = (tt < LS) ? syb[b * LS + tt] : qf[b * LQ + (tt - LS)];
  X0[t] = emb[(size_t)id * DG + c];
}

__global__ void k_add_pos(float* __restrict__ x, const float* __restrict__ pos) {
  int t = blockIdx.x * blockDim.x + threadIdx.x;
  if (t >= BT * HH) return;
  int row = t >> 9, c = t & 511;
  x[t] += pos[(row & 255) * HH + c];
}

// build graph_diag & graph masks [B,T,T]
__global__ void k_masks(const float* __restrict__ qm, const float* __restrict__ sg,
                        const float* __restrict__ qg,
                        float* __restrict__ gd, float* __restrict__ gg) {
  long long t = (long long)blockIdx.x * blockDim.x + threadIdx.x;
  if (t >= (long long)BB * TT * TT) return;
  int j = (int)(t & 255);
  long long r = t >> 8;
  int i = (int)(r & 255);
  int b = (int)(r >> 8);
  float gdv, ggv;
  if (i < LS && j < LS) {
    gdv = 0.f;
    ggv = sg[((size_t)b * LS + i) * LS + j];
  } else if (i >= LS && j >= LS) {
    gdv = qm[((size_t)b * LQ + (i - LS)) * LQ + (j - LS)];
    ggv = qg[((size_t)b * LQ + (i - LS)) * LQ + (j - LS)];
  } else {
    gdv = 0.f; ggv = 1.f;   // 1 - mask, mask==0 off-diagonal
  }
  gd[t] = gdv; gg[t] = ggv;
}

__global__ void k_dec_init(const float* __restrict__ dec_emb,
                           const float* __restrict__ dec_pos, float* __restrict__ dec) {
  int t = blockIdx.x * blockDim.x + threadIdx.x;
  if (t >= BB * HH) return;
  int c = t & 511;
  dec[t] = dec_emb[2 * HH + c] * 22.62741699796952f + dec_pos[c];
}

// decoder self-attention, T=1: softmax(1x1)=1 -> out = q_mask * V
__global__ __launch_bounds__(256) void k_dec_self(const float* __restrict__ dec,
                                                  const float* __restrict__ Vd,
                                                  float* __restrict__ dt) {
  __shared__ float red[256];
  int b = blockIdx.x, t = threadIdx.x;
  red[t] = dec[b * HH + t] + dec[b * HH + 256 + t];
  __syncthreads();
  for (int s = 128; s > 0; s >>= 1) { if (t < s) red[t] += red[t + s]; __syncthreads(); }
  float qm = (fabsf(red[0]) != 0.f) ? 1.f : 0.f;
  dt[b * HH + t]       = qm * Vd[b * HH + t];
  dt[b * HH + 256 + t] = qm * Vd[b * HH + 256 + t];
}

// decoder cross-attention: 1 query x 256 keys per (b,h)
__global__ __launch_bounds__(256) void k_cross(
    const float* __restrict__ Qd, const float* __restrict__ Kx,
    const float* __restrict__ Vx, const float* __restrict__ kmx,
    const float* __restrict__ dec, float* __restrict__ dt) {
  __shared__ float qv[HD];
  __shared__ float p[TT];
  __shared__ float red[256];
  __shared__ float qmsh;
  int bid = blockIdx.x, h = bid & 7, b = bid >> 3, t = threadIdx.x;
  int c0 = h * HD;
  red[t] = dec[b * HH + t] + dec[b * HH + 256 + t];
  if (t < HD) qv[t] = Qd[b * HH + c0 + t];
  __syncthreads();
  for (int s = 128; s > 0; s >>= 1) { if (t < s) red[t] += red[t + s]; __syncthreads(); }
  if (t == 0) qmsh = (fabsf(red[0]) != 0.f) ? 1.f : 0.f;
  __syncthreads();
  float dot = 0.f;
  const float* kr = Kx + ((size_t)b * TT + t) * HH + c0;
  for (int c = 0; c < HD; c++) dot += qv[c] * kr[c];
  float s = dot * 0.125f;
  if (kmx[b * TT + t] == 0.f) s = NEGV;
  p[t] = s; red[t] = s;
  __syncthreads();
  for (int st = 128; st > 0; st >>= 1) { if (t < st) red[t] = fmaxf(red[t], red[t + st]); __syncthreads(); }
  float m = red[0];
  __syncthreads();
  float e = __expf(p[t] - m);
  p[t] = e; red[t] = e;
  __syncthreads();
  for (int st = 128; st > 0; st >>= 1) { if (t < st) red[t] += red[t + st]; __syncthreads(); }
  float inv = qmsh / red[0];
  __syncthreads();
  int c = t & 63, part = t >> 6;
  float acc = 0.f;
  for (int k = part * 64; k < part * 64 + 64; k++)
    acc += p[k] * Vx[((size_t)b * TT + k) * HH + c0 + c];
  red[t] = acc;
  __syncthreads();
  if (t < HD)
    dt[b * HH + c0 + t] = (red[t] + red[t + 64] + red[t + 128] + red[t + 192]) * inv;
}

// ---------------------------------------------------------------------------
extern "C" void kernel_launch(void* const* d_in, const int* in_sizes, int n_in,
                              void* d_out, int out_size, void* d_ws, size_t ws_size,
                              hipStream_t stream) {
  (void)in_sizes; (void)n_in; (void)out_size; (void)ws_size;
  const int*   syb_ipt = (const int*)d_in[0];
  const int*   q_fea   = (const int*)d_in[1];
  const float* q_mask  = (const float*)d_in[3];
  const float* syb_g   = (const float*)d_in[4];
  const float* q_g     = (const float*)d_in[5];
  const float* emb     = (const float*)d_in[6];
  const float* mlp_w1  = (const float*)d_in[7];
  const float* mlp_b1  = (const float*)d_in[8];
  const float* mlp_w2  = (const float*)d_in[9];
  const float* mlp_b2  = (const float*)d_in[10];
  const float* pos_enc = (const float*)d_in[11];
  const float* enc_wq  = (const float*)d_in[12];
  const float* enc_bq  = (const float*)d_in[13];
  const float* enc_wk  = (const float*)d_in[14];
  const float* enc_bk  = (const float*)d_in[15];
  const float* enc_wv  = (const float*)d_in[16];
  const float* enc_bv  = (const float*)d_in[17];
  const float* enc_ag  = (const float*)d_in[18];
  const float* enc_ab  = (const float*)d_in[19];
  const float* enc_fw1 = (const float*)d_in[20];
  const float* enc_fb1 = (const float*)d_in[21];
  const float* enc_fw2 = (const float*)d_in[22];
  const float* enc_fb2 = (const float*)d_in[23];
  const float* enc_fg  = (const float*)d_in[24];
  const float* enc_fbt = (const float*)d_in[25];
  const float* dec_emb = (const float*)d_in[26];
  const float* dec_pos = (const float*)d_in[27];
  const float* dsa_wv  = (const float*)d_in[32];
  const float* dsa_bv  = (const float*)d_in[33];
  const float* dsa_g   = (const float*)d_in[34];
  const float* dsa_bt  = (const float*)d_in[35];
  const float* dva_wq  = (const float*)d_in[36];
  const float* dva_bq  = (const float*)d_in[37];
  const float* dva_wk  = (const float*)d_in[38];
  const float* dva_bk  = (const float*)d_in[39];
  const float* dva_wv  = (const float*)d_in[40];
  const float* dva_bv  = (const float*)d_in[41];
  const float* dva_g   = (const float*)d_in[42];
  const float* dva_bt  = (const float*)d_in[43];
  const float* dff_w1  = (const float*)d_in[44];
  const float* dff_b1  = (const float*)d_in[45];
  const float* dff_w2  = (const float*)d_in[46];
  const float* dff_b2  = (const float*)d_in[47];
  const float* dff_g   = (const float*)d_in[48];
  const float* dff_bt  = (const float*)d_in[49];

  // workspace layout (floats)
  float* w = (float*)d_ws;
  size_t off = 0;
  float* x   = w + off; off += (size_t)BT * HH;       // 8192x512
  float* tb  = w + off; off += (size_t)BT * HH;       // tmp (pre-LN)
  float* qb  = w + off; off += (size_t)BT * HH;
  float* kb  = w + off; off += (size_t)BT * HH;
  float* vb  = w + off; off += (size_t)BT * HH;
  float* hb  = w + off; off += (size_t)BT * DM;       // 8192x2048 hidden
  float* x0  = w + off; off += (size_t)BT * DG;       // 8192x300
  float* gd  = w + off; off += (size_t)BB * TT * TT;  // graph_diag
  float* gg  = w + off; off += (size_t)BB * TT * TT;  // graph
  float* km  = w + off; off += BT;
  float* dec = w + off; off += BB * HH;
  float* dt  = w + off; off += BB * HH;
  float* dq  = w + off; off += BB * HH;
  float* dv  = w + off; off += BB * HH;

  auto gemm = [&](const float* A, int lda, const float* W, const float* bias,
                  float* C, int M, int N, int K, int act) {
    dim3 g((M + GBM - 1) / GBM, N / GBN);
    k_gemm_nn<<<g, dim3(256), 0, stream>>>(A, lda, W, bias, C, M, N, K, act);
  };

  // ---- front end -------------------------------------------------------
  k_gather<<<(BT * DG + 255) / 256, 256, 0, stream>>>(syb_ipt, q_fea, emb, x0);
  gemm(x0, DG, mlp_w1, mlp_b1, hb, BT, DM, DG, 1);
  gemm(hb, DM, mlp_w2, mlp_b2, x, BT, HH, DM, 0);
  k_add_pos<<<(BT * HH + 255) / 256, 256, 0, stream>>>(x, pos_enc);
  k_masks<<<(BB * TT * TT + 255) / 256, 256, 0, stream>>>(q_mask, syb_g, q_g, gd, gg);

  // ---- encoder ---------------------------------------------------------
  for (int i = 0; i < NB; i++) {
    const float* graph = (i < 2) ? gd : gg;
    k_rowmask<<<BT / 8, 256, 0, stream>>>(x, km, BT);
    gemm(x, HH, enc_wq + (size_t)i * HH * HH, enc_bq + i * HH, qb, BT, HH, HH, 1);
    gemm(x, HH, enc_wk + (size_t)i * HH * HH, enc_bk + i * HH, kb, BT, HH, HH, 1);
    gemm(x, HH, enc_wv + (size_t)i * HH * HH, enc_bv + i * HH, vb, BT, HH, HH, 1);
    k_attn_enc<<<BB * NHD * 8, 64, 0, stream>>>(qb, kb, vb, km, graph, tb);
    k_add_ln<<<BT / 8, 256, 0, stream>>>(tb, x, enc_ag + i * HH, enc_ab + i * HH, x, BT);
    gemm(x, HH, enc_fw1 + (size_t)i * HH * HF, enc_fb1 + i * HF, hb, BT, HF, HH, 1);
    gemm(hb, HF, enc_fw2 + (size_t)i * HF * HH, enc_fb2 + i * HH, tb, BT, HH, HF, 0);
    k_add_ln<<<BT / 8, 256, 0, stream>>>(tb, x, enc_fg + i * HH, enc_fbt + i * HH, x, BT);
  }

  // key mask over final encoder output (fixed across decoder blocks)
  k_rowmask<<<BT / 8, 256, 0, stream>>>(x, km, BT);

  // ---- decoder ---------------------------------------------------------
  k_dec_init<<<(BB * HH + 255) / 256, 256, 0, stream>>>(dec_emb, dec_pos, dec);
  for (int i = 0; i < NB; i++) {
    // self-attn (T=1 shortcut): out = q_mask * relu(dec@wv+bv)
    gemm(dec, HH, dsa_wv + (size_t)i * HH * HH, dsa_bv + i * HH, dv, BB, HH, HH, 1);
    k_dec_self<<<BB, 256, 0, stream>>>(dec, dv, dt);
    k_add_ln<<<(BB + 7) / 8, 256, 0, stream>>>(dt, dec, dsa_g + i * HH, dsa_bt + i * HH, dec, BB);
    // cross-attn
    gemm(dec, HH, dva_wq + (size_t)i * HH * HH, dva_bq + i * HH, dq, BB, HH, HH, 1);
    gemm(x, HH, dva_wk + (size_t)i * HH * HH, dva_bk + i * HH, kb, BT, HH, HH, 1);
    gemm(x, HH, dva_wv + (size_t)i * HH * HH, dva_bv + i * HH, vb, BT, HH, HH, 1);
    k_cross<<<BB * NHD, 256, 0, stream>>>(dq, kb, vb, km, dec, dt);
    k_add_ln<<<(BB + 7) / 8, 256, 0, stream>>>(dt, dec, dva_g + i * HH, dva_bt + i * HH, dec, BB);
    // ffn
    gemm(dec, HH, dff_w1 + (size_t)i * HH * HF, dff_b1 + i * HF, hb, BB, HF, HH, 1);
    gemm(hb, HF, dff_w2 + (size_t)i * HF * HH, dff_b2 + i * HH, dt, BB, HH, HF, 0);
    k_add_ln<<<(BB + 7) / 8, 256, 0, stream>>>(dt, dec, dff_g + i * HH, dff_bt + i * HH, dec, BB);
  }

  hipMemcpyAsync(d_out, dec, (size_t)BB * HH * sizeof(float),
                 hipMemcpyDeviceToDevice, stream);
}
